// GAT_FOR_HETERO_10368051052622
// MI455X (gfx1250) — compile-verified, matrix operand order checked by reading, not run on hardware
//
#include <hip/hip_runtime.h>

// ---------------- problem constants (match reference) ----------------
#define NN     50000      // nodes
#define NE     800000     // edges
#define DF     128        // feature dim (D_IN == HEADS*CH)
#define NH     4          // heads
#define NC     32         // channels per head
#define NLAY   3
#define NEG_SLOPE 0.2f
#define MT     5          // row-tiles per wave (3125 % 5 == 0)

typedef __attribute__((ext_vector_type(16))) _Float16 v16h;
typedef __attribute__((ext_vector_type(8)))  _Float16 v8h;
typedef __attribute__((ext_vector_type(8)))  float    v8f;

// ---- order-preserving float <-> uint encoding for atomicMax on floats ----
__device__ __forceinline__ unsigned encf(float f) {
    unsigned u = __float_as_uint(f);
    return (u & 0x80000000u) ? ~u : (u | 0x80000000u);
}
__device__ __forceinline__ float decf(unsigned e) {
    unsigned u = (e & 0x80000000u) ? (e & 0x7FFFFFFFu) : ~e;
    return __uint_as_float(u);
}

// ---------------------------------------------------------------------
// 0) zero-fill
// ---------------------------------------------------------------------
__global__ void gat_zero(float* __restrict__ p, int n) {
    int i = blockIdx.x * blockDim.x + threadIdx.x;
    if (i < n) p[i] = 0.0f;
}

// ---------------------------------------------------------------------
// 0b) f32 -> f16 convert (4 elems/thread, vectorized)
// ---------------------------------------------------------------------
__global__ void gat_cvt_f16(const float* __restrict__ in,
                            _Float16* __restrict__ out, int n4) {
    int i = blockIdx.x * blockDim.x + threadIdx.x;
    if (i >= n4) return;
    float4 v = ((const float4*)in)[i];
    __attribute__((ext_vector_type(4))) _Float16 h =
        {(_Float16)v.x, (_Float16)v.y, (_Float16)v.z, (_Float16)v.w};
    *(__attribute__((ext_vector_type(4))) _Float16*)(out + (size_t)i * 4) = h;
}

// ---------------------------------------------------------------------
// 0c) Pack Wsrc/Wdst into the exact per-lane WMMA B-operand image:
//     wpack[((ct*4 + ks)*32 + lane)*16 + j] = (f16) W[(kbase+kmap(j))*DF + col]
//     ct in [0,16): 0-7 -> Wsrc col-tiles, 8-15 -> Wdst col-tiles.
// ---------------------------------------------------------------------
__global__ void gat_packW(const float* __restrict__ Wsrc,
                          const float* __restrict__ Wdst,
                          _Float16* __restrict__ wpack) {
    int idx = blockIdx.x * blockDim.x + threadIdx.x;   // < 16*4*32 = 2048
    if (idx >= 2048) return;
    int lane = idx & 31;
    int ks   = (idx >> 5) & 3;
    int ct   = idx >> 7;                    // 0..15
    const float* W = (ct < 8) ? Wsrc : Wdst;
    int col  = (ct & 7) * 16 + (lane & 15);
    int hi   = lane >> 4;
    int kbase = ks * 32;
    v16h b;
#pragma unroll
    for (int j = 0; j < 8; ++j) {
        b[j]     = (_Float16)W[(size_t)(kbase + hi * 8 + j) * DF + col];
        b[j + 8] = (_Float16)W[(size_t)(kbase + 16 + hi * 8 + j) * DF + col];
    }
    *(v16h*)(wpack + (size_t)idx * 16) = b;
}

// ---------------------------------------------------------------------
// 1) Dual projection via WMMA:  xs = xh @ Wsrc,  xd = xh @ Wdst
//    One wave32 computes MT=5 row-tiles of one 16-col tile; K=128 in
//    4 steps of 32.  Per K-step: B loaded once (reused 5x) and ALL 5
//    A tiles loaded into distinct registers BEFORE the WMMA chain, so
//    the scheduler can use partial loadcnt waits and pipeline the next
//    K-step's loads under the matrix ops.
// ---------------------------------------------------------------------
__global__ __launch_bounds__(256)
void gat_proj_wmma(const _Float16* __restrict__ xh,
                   const _Float16* __restrict__ wpack,
                   float* __restrict__ xs,
                   float* __restrict__ xd)
{
    const int wave = threadIdx.x >> 5;
    const int lane = threadIdx.x & 31;
    const int tile = blockIdx.x * 8 + wave;      // < 10000
    const int rowBlk = tile >> 4;                // 0..624
    const int ct     = tile & 15;
    float* __restrict__ out = (ct < 8) ? xs : xd;
    const int col0 = (ct & 7) * 16;
    const int row0 = rowBlk * (16 * MT);

    const int r  = lane & 15;      // A row / D col within tile
    const int hi = lane >> 4;      // K half selector

    const _Float16* __restrict__ xrow = xh + (size_t)(row0 + r) * DF;
    const _Float16* __restrict__ bp   = wpack + ((size_t)ct * 4 * 32 + lane) * 16;

    v8f acc[MT] = {};
#pragma unroll
    for (int ks = 0; ks < 4; ++ks) {
        const int kbase = ks * 32;
        // B for this K-step (reused by all MT row-tiles)
        const v16h b = *(const v16h*)(bp + (size_t)ks * 32 * 16);
        // batch-load all MT A tiles into distinct registers
        v16h a[MT];
#pragma unroll
        for (int m = 0; m < MT; ++m) {
            const _Float16* ar = xrow + (size_t)m * 16 * DF;
            v8h alo = *(const v8h*)(ar + kbase + hi * 8);
            v8h ahi = *(const v8h*)(ar + kbase + 16 + hi * 8);
#pragma unroll
            for (int j = 0; j < 8; ++j) { a[m][j] = alo[j]; a[m][j + 8] = ahi[j]; }
        }
        // WMMA chain (no shared A registers -> partial waits possible)
#pragma unroll
        for (int m = 0; m < MT; ++m) {
            acc[m] = __builtin_amdgcn_wmma_f32_16x16x32_f16(
                         false, a[m], false, b, (short)0, acc[m], false, false);
        }
    }
    // D tile m: VGPR i -> row (row0 + m*16 + hi*8 + i), col (col0+r)
#pragma unroll
    for (int m = 0; m < MT; ++m) {
#pragma unroll
        for (int i = 0; i < 8; ++i) {
            out[(size_t)(row0 + m * 16 + hi * 8 + i) * DF + col0 + r] = acc[m][i];
        }
    }
}

// ---------------------------------------------------------------------
// 2) Per-node attention logits: al_s[n,h] = <xs[n,h,:], a_src[h,:]>
// ---------------------------------------------------------------------
__global__ void gat_alpha(const float* __restrict__ xs,
                          const float* __restrict__ xd,
                          const float* __restrict__ a_src,
                          const float* __restrict__ a_dst,
                          float* __restrict__ al_s,
                          float* __restrict__ al_d)
{
    int i = blockIdx.x * blockDim.x + threadIdx.x;   // node*NH + h
    if (i >= NN * NH) return;
    int h = i & 3, n = i >> 2;
    const float4* ps = (const float4*)(xs + (size_t)n * DF + h * NC);
    const float4* pd = (const float4*)(xd + (size_t)n * DF + h * NC);
    const float4* as = (const float4*)(a_src + h * NC);
    const float4* ad = (const float4*)(a_dst + h * NC);
    float ss = 0.f, sd = 0.f;
#pragma unroll
    for (int j = 0; j < 8; ++j) {
        float4 v = ps[j], w = as[j];
        ss += v.x * w.x + v.y * w.y + v.z * w.z + v.w * w.w;
        float4 u = pd[j], t = ad[j];
        sd += u.x * t.x + u.y * t.y + u.z * t.z + u.w * t.w;
    }
    al_s[i] = ss;
    al_d[i] = sd;
}

// ---------------------------------------------------------------------
// 3) Edge pass A: e = leaky_relu(al_s[src]+al_d[dst]); segment max via
//    order-preserving uint atomicMax.
// ---------------------------------------------------------------------
__global__ void gat_edge_max(const int* __restrict__ src,
                             const int* __restrict__ dst,
                             const float* __restrict__ al_s,
                             const float* __restrict__ al_d,
                             float* __restrict__ ebuf,
                             unsigned* __restrict__ mbuf)
{
    int i = blockIdx.x * blockDim.x + threadIdx.x;   // edge*NH + h
    if (i >= NE * NH) return;
    int h = i & 3, e = i >> 2;
    int s = src[e], d = dst[e];
    float v = al_s[s * NH + h] + al_d[d * NH + h];
    v = (v > 0.f) ? v : NEG_SLOPE * v;
    ebuf[i] = v;
    atomicMax(&mbuf[d * NH + h], encf(v));
}

// ---------------------------------------------------------------------
// 4) Edge pass B: ex = exp(e - m[dst]); segment sum.
// ---------------------------------------------------------------------
__global__ void gat_edge_exp(const int* __restrict__ dst,
                             float* __restrict__ ebuf,
                             const unsigned* __restrict__ mbuf,
                             float* __restrict__ sbuf)
{
    int i = blockIdx.x * blockDim.x + threadIdx.x;   // edge*NH + h
    if (i >= NE * NH) return;
    int h = i & 3, e = i >> 2;
    int d = dst[e];
    float m  = decf(mbuf[d * NH + h]);
    float ex = __expf(ebuf[i] - m);
    ebuf[i] = ex;
    unsafeAtomicAdd(&sbuf[d * NH + h], ex);
}

// ---------------------------------------------------------------------
// 5) Edge pass C: out[dst] += xs[src] * alpha.  One wave32 per edge;
//    lane owns 4 contiguous channels (head = lane/8).
// ---------------------------------------------------------------------
__global__ void gat_edge_agg(const int* __restrict__ src,
                             const int* __restrict__ dst,
                             const float* __restrict__ xs,
                             const float* __restrict__ ebuf,
                             const float* __restrict__ sbuf,
                             float* __restrict__ acc)
{
    long long gid = (long long)blockIdx.x * blockDim.x + threadIdx.x;
    if (gid >= (long long)NE * 32) return;
    int lane = (int)(gid & 31);
    int e    = (int)(gid >> 5);
    int s = src[e], d = dst[e];
    int h = lane >> 3;                          // (lane*4)/32
    float alpha = ebuf[e * NH + h] / (sbuf[d * NH + h] + 1e-16f);
    const float4 v = *(const float4*)(xs + (size_t)s * DF + lane * 4);
    float* o = acc + (size_t)d * DF + lane * 4;
    unsafeAtomicAdd(o + 0, v.x * alpha);
    unsafeAtomicAdd(o + 1, v.y * alpha);
    unsafeAtomicAdd(o + 2, v.z * alpha);
    unsafeAtomicAdd(o + 3, v.w * alpha);
}

// ---------------------------------------------------------------------
// 6) bias (+ optional relu), write f32 layer output AND f16 copy for the
//    next layer's WMMA A-operand.
// ---------------------------------------------------------------------
__global__ void gat_finish(const float* __restrict__ acc,
                           const float* __restrict__ bias,
                           float* __restrict__ dstp,
                           _Float16* __restrict__ dsth, int doRelu)
{
    int i = blockIdx.x * blockDim.x + threadIdx.x;
    if (i >= NN * DF) return;
    float v = acc[i] + bias[i & (DF - 1)];
    if (doRelu) v = fmaxf(v, 0.f);
    dstp[i] = v;
    dsth[i] = (_Float16)v;
}

// ---------------------------------------------------------------------
extern "C" void kernel_launch(void* const* d_in, const int* in_sizes, int n_in,
                              void* d_out, int out_size, void* d_ws, size_t ws_size,
                              hipStream_t stream)
{
    const float* x0      = (const float*)d_in[0];
    const int*   edge    = (const int*)d_in[1];
    const int*   src     = edge;
    const int*   dst     = edge + NE;
    const float* Wsrc    = (const float*)d_in[2];   // [3,128,128]
    const float* Wdst    = (const float*)d_in[3];
    const float* att_src = (const float*)d_in[4];   // [3,4,32]
    const float* att_dst = (const float*)d_in[5];
    const float* bias    = (const float*)d_in[6];   // [3,128]
    float* out = (float*)d_out;

    // ---- workspace carve-up ----
    char* ws = (char*)d_ws;
    const size_t szNF  = (size_t)NN * DF * sizeof(float);      // 25.6 MB
    const size_t szNH  = (size_t)NN * NH * sizeof(float);
    const size_t szEH  = (size_t)NE * NH * sizeof(float);
    const size_t szNFh = (size_t)NN * DF * sizeof(_Float16);   // 12.8 MB
    const size_t szWP  = (size_t)2048 * 16 * sizeof(_Float16); // 64 KB
    float*    xs    = (float*)ws;             ws += szNF;
    float*    xd    = (float*)ws;             ws += szNF;
    float*    acc   = (float*)ws;             ws += szNF;
    float*    ebuf  = (float*)ws;             ws += szEH;
    float*    al_s  = (float*)ws;             ws += szNH;
    float*    al_d  = (float*)ws;             ws += szNH;
    float*    sbuf  = (float*)ws;             ws += szNH;
    unsigned* mbuf  = (unsigned*)ws;          ws += szNH;
    float*    xcur  = (float*)ws;             ws += szNF;
    _Float16* xh    = (_Float16*)ws;          ws += szNFh;
    _Float16* wpack = (_Float16*)ws;          ws += szWP;
    (void)ws_size; (void)in_sizes; (void)n_in; (void)out_size;

    const int B = 256;

    // layer-0 A-operand in f16
    gat_cvt_f16<<<(NN * DF / 4 + B - 1) / B, B, 0, stream>>>(x0, xh, NN * DF / 4);

    for (int L = 0; L < NLAY; ++L) {
        // zero per-layer accumulators (enc lower bound == 0 bit pattern)
        gat_zero<<<(NN * DF + B - 1) / B, B, 0, stream>>>(acc, NN * DF);
        gat_zero<<<(NN * NH + B - 1) / B, B, 0, stream>>>(sbuf, NN * NH);
        gat_zero<<<(NN * NH + B - 1) / B, B, 0, stream>>>((float*)mbuf, NN * NH);

        gat_packW<<<8, 256, 0, stream>>>(Wsrc + (size_t)L * DF * DF,
                                         Wdst + (size_t)L * DF * DF, wpack);

        // 625 row-blocks * 16 col-tiles = 10000 waves / 8 = 1250 blocks
        gat_proj_wmma<<<1250, 256, 0, stream>>>(xh, wpack, xs, xd);

        gat_alpha<<<(NN * NH + B - 1) / B, B, 0, stream>>>(xs, xd,
                                                           att_src + L * NH * NC,
                                                           att_dst + L * NH * NC,
                                                           al_s, al_d);
        gat_edge_max<<<(NE * NH) / B, B, 0, stream>>>(src, dst, al_s, al_d, ebuf, mbuf);
        gat_edge_exp<<<(NE * NH) / B, B, 0, stream>>>(dst, ebuf, mbuf, sbuf);
        gat_edge_agg<<<((long long)NE * 32) / B, B, 0, stream>>>(src, dst, xs, ebuf, sbuf, acc);

        float* dest = (L == NLAY - 1) ? out : xcur;
        gat_finish<<<(NN * DF) / B, B, 0, stream>>>(acc, bias + L * DF, dest, xh,
                                                    (L < NLAY - 1) ? 1 : 0);
    }
}
